// GAT_79998060855859
// MI455X (gfx1250) — compile-verified, matrix operand order checked by reading, not run on hardware
//
#include <hip/hip_runtime.h>
#include <math.h>

// ---------------------------------------------------------------------------
// GAT (3-layer) on gfx1250: LDS-staged bf16 WMMA GEMMs + CSR gather attention.
// ---------------------------------------------------------------------------

typedef __attribute__((ext_vector_type(16))) __bf16 v16bf;
typedef __attribute__((ext_vector_type(8)))  __bf16 v8bf;
typedef __attribute__((ext_vector_type(8)))  float  v8f;

#define NEG_SLOPE 0.2f
#define BN_EPS 1e-5f
#define SCAN_B 256

// ---------------- fills -----------------------------------------------------
__global__ void fill_f32_kernel(float* p, float v, size_t n) {
    size_t t = (size_t)blockIdx.x * blockDim.x + threadIdx.x;
    size_t stride = (size_t)gridDim.x * blockDim.x;
    for (; t < n; t += stride) p[t] = v;
}
__global__ void fill_i32_kernel(int* p, int v, size_t n) {
    size_t t = (size_t)blockIdx.x * blockDim.x + threadIdx.x;
    size_t stride = (size_t)gridDim.x * blockDim.x;
    for (; t < n; t += stride) p[t] = v;
}

// ---------------- conversions ----------------------------------------------
__global__ void f32_to_bf16_kernel(const float* __restrict__ in, __bf16* __restrict__ out, size_t n) {
    size_t t = (size_t)blockIdx.x * blockDim.x + threadIdx.x;
    size_t stride = (size_t)gridDim.x * blockDim.x;
    for (; t < n; t += stride) out[t] = (__bf16)in[t];
}
// W: [K, D] row-major fp32 -> Wt: [D, K] bf16 (so B fragments load contiguously)
__global__ void transpose_to_bf16_kernel(const float* __restrict__ W, __bf16* __restrict__ Wt,
                                         int K, int D) {
    size_t t = (size_t)blockIdx.x * blockDim.x + threadIdx.x;
    size_t n = (size_t)K * D;
    if (t >= n) return;
    int k = (int)(t / D);
    int d = (int)(t % D);
    Wt[(size_t)d * K + k] = (__bf16)W[t];
}

// ---------------- WMMA GEMM: C[M,Dout] = A[M,K] (bf16) * Bt[Dout,K]^T -------
// Block: 8 waves, 128 rows x 64 cols. B strip (64 x K) staged once in LDS.
#define GEMM_WAVES 8
#define KMAX 256
#define BPAD 8                      // 16B pad -> conflict-free ds_load_b128
__global__ void wmma_gemm_kernel(const __bf16* __restrict__ A, const __bf16* __restrict__ Bt,
                                 float* __restrict__ C, int M, int K, int Dout) {
    __shared__ __bf16 sB[64 * (KMAX + BPAD)];
    const int wave = threadIdx.x >> 5;
    const int lane = threadIdx.x & 31;
    const int mtile = blockIdx.x * GEMM_WAVES + wave;
    const int row0 = mtile * 16;
    const int col0 = blockIdx.y * 64;
    const int ldb = K + BPAD;

    // ---- cooperative stage of the 64-column B strip into LDS (b128 chunks) ----
    {
        const int chunks_per_col = K >> 3;                 // v8bf chunks per column
        const int total = 64 * chunks_per_col;
        const v8bf* gsrc = (const v8bf*)(Bt + (size_t)col0 * K);
        for (int idx = threadIdx.x; idx < total; idx += blockDim.x) {
            int c = idx / chunks_per_col;
            int j = idx - c * chunks_per_col;
            *(v8bf*)(sB + c * ldb + j * 8) = gsrc[(size_t)c * chunks_per_col + j];
        }
    }
    __syncthreads();
    if (row0 >= M) return;                                  // tail waves retire after barrier

    const int half = lane >> 4;          // 0: lanes 0-15, 1: lanes 16-31
    const int l16  = lane & 15;
    const int klo  = half * 8;           // per-ISA A/B fragment K base for this half-wave

    const __bf16* arow = A + (size_t)(row0 + l16) * K + klo;

    v8f acc[4];
    #pragma unroll
    for (int i = 0; i < 4; ++i) acc[i] = (v8f){};

    for (int k = 0; k < K; k += 32) {
        v8bf alo = *(const v8bf*)(arow + k);        // K = klo .. klo+7
        v8bf ahi = *(const v8bf*)(arow + k + 16);   // K = klo+16 .. klo+23
        v16bf afrag = __builtin_shufflevector(alo, ahi,
            0,1,2,3,4,5,6,7,8,9,10,11,12,13,14,15);
        #pragma unroll
        for (int nt = 0; nt < 4; ++nt) {
            const __bf16* brow = sB + (nt * 16 + l16) * ldb + klo + k;
            v8bf blo = *(const v8bf*)(brow);
            v8bf bhi = *(const v8bf*)(brow + 16);
            v16bf bfrag = __builtin_shufflevector(blo, bhi,
                0,1,2,3,4,5,6,7,8,9,10,11,12,13,14,15);
            acc[nt] = __builtin_amdgcn_wmma_f32_16x16x32_bf16(
                false, afrag, false, bfrag, (short)0, acc[nt], false, false);
        }
    }

    // C/D layout: row = r + 8*half, col = lane%16
    float* cbase = C + (size_t)(row0 + half * 8) * Dout + col0 + l16;
    #pragma unroll
    for (int r = 0; r < 8; ++r) {
        cbase[(size_t)r * Dout +  0] = acc[0][r];
        cbase[(size_t)r * Dout + 16] = acc[1][r];
        cbase[(size_t)r * Dout + 32] = acc[2][r];
        cbase[(size_t)r * Dout + 48] = acc[3][r];
    }
}

// ---------------- per-node attention logits (alpha_src / alpha_dst) ---------
// feat: [N, D]; a_s, a_d: flat [D] (head-major). blockDim.x == D, one node/block.
__global__ void alpha_kernel(const float* __restrict__ feat,
                             const float* __restrict__ a_s, const float* __restrict__ a_d,
                             float* __restrict__ asrc, float* __restrict__ adst,
                             int D, int C) {
    __shared__ float ss[256];
    __shared__ float sd[256];
    const int n = blockIdx.x;
    const int t = threadIdx.x;
    float v = feat[(size_t)n * D + t];
    ss[t] = v * a_s[t];
    sd[t] = v * a_d[t];
    __syncthreads();
    for (int s = C >> 1; s >= 1; s >>= 1) {
        if ((t & (C - 1)) < s) { ss[t] += ss[t + s]; sd[t] += sd[t + s]; }
        __syncthreads();
    }
    if ((t & (C - 1)) == 0) {
        int hd = t / C;
        int H = D / C;
        asrc[(size_t)n * H + hd] = ss[t];
        adst[(size_t)n * H + hd] = sd[t];
    }
}

// ================== CSR construction (dst is layer-invariant) ===============
__global__ void deg_count_kernel(const int* __restrict__ ei, int E, int N, int* __restrict__ deg) {
    int e = blockIdx.x * blockDim.x + threadIdx.x;
    if (e >= E + N) return;
    int dst = (e < E) ? ei[E + e] : (e - E);
    atomicAdd(&deg[dst], 1);
}

// block-local exclusive scan; emits per-block inclusive total
__global__ void scan_block_kernel(const int* __restrict__ deg, int* __restrict__ rowptr,
                                  int* __restrict__ blocksum, int N) {
    __shared__ int s[SCAN_B];
    const int t = threadIdx.x;
    const int i = blockIdx.x * SCAN_B + t;
    int v = (i < N) ? deg[i] : 0;
    s[t] = v;
    __syncthreads();
    for (int o = 1; o < SCAN_B; o <<= 1) {
        int add = (t >= o) ? s[t - o] : 0;
        __syncthreads();
        s[t] += add;
        __syncthreads();
    }
    if (i < N) rowptr[i] = s[t] - v;                 // exclusive within block
    if (t == SCAN_B - 1) blocksum[blockIdx.x] = s[t];
}

// single-block exclusive scan of block sums (nb <= 256)
__global__ void scan_sums_kernel(int* __restrict__ blocksum, int nb) {
    __shared__ int s[SCAN_B];
    const int t = threadIdx.x;
    int v = (t < nb) ? blocksum[t] : 0;
    s[t] = v;
    __syncthreads();
    for (int o = 1; o < SCAN_B; o <<= 1) {
        int add = (t >= o) ? s[t - o] : 0;
        __syncthreads();
        s[t] += add;
        __syncthreads();
    }
    if (t < nb) blocksum[t] = s[t] - v;              // exclusive block offsets
}

__global__ void scan_add_kernel(int* __restrict__ rowptr, const int* __restrict__ blockoff, int N) {
    int i = blockIdx.x * blockDim.x + threadIdx.x;
    if (i < N) rowptr[i] += blockoff[i / SCAN_B];
}

__global__ void csr_scatter_kernel(const int* __restrict__ ei, int E, int N,
                                   const int* __restrict__ rowptr, int* __restrict__ fillp,
                                   int* __restrict__ eidx) {
    int e = blockIdx.x * blockDim.x + threadIdx.x;
    if (e >= E + N) return;
    int dst = (e < E) ? ei[E + e] : (e - E);
    int pos = rowptr[dst] + atomicAdd(&fillp[dst], 1);
    eidx[pos] = e;
}

// ========== fused segment-softmax + weighted gather (no atomics) ============
// one block per destination node, blockDim.x == D (= H*C)
__global__ void gat_aggregate_csr_kernel(const int* __restrict__ ei, int E,
                                         const int* __restrict__ rowptr, const int* __restrict__ deg,
                                         const int* __restrict__ eidx,
                                         const float* __restrict__ feat,
                                         const float* __restrict__ asrc, const float* __restrict__ adst,
                                         float* __restrict__ agg, int D, int C, int H) {
    __shared__ float red[256];
    __shared__ float shead[8];                 // [0..3]=max, [4..7]=denom
    const int n = blockIdx.x;
    const int t = threadIdx.x;
    const int bd = blockDim.x;
    const int start = rowptr[n];
    const int dcount = deg[n];

    float ad_loc[4];
    for (int h = 0; h < H; ++h) ad_loc[h] = adst[(size_t)n * H + h];

    // ---- phase A: per-head max over incident edges ----
    float pmax[4];
    for (int h = 0; h < H; ++h) pmax[h] = -INFINITY;
    for (int i = t; i < dcount; i += bd) {
        int e = eidx[start + i];
        int src = (e < E) ? ei[e] : (e - E);
        for (int h = 0; h < H; ++h) {
            float s = asrc[(size_t)src * H + h] + ad_loc[h];
            s = (s > 0.f) ? s : NEG_SLOPE * s;
            pmax[h] = fmaxf(pmax[h], s);
        }
    }
    for (int h = 0; h < H; ++h) {
        red[t] = pmax[h];
        __syncthreads();
        for (int s2 = bd >> 1; s2 >= 1; s2 >>= 1) {
            if (t < s2) red[t] = fmaxf(red[t], red[t + s2]);
            __syncthreads();
        }
        if (t == 0) shead[h] = red[0];
        __syncthreads();
    }

    // ---- phase B: per-head denom = sum(exp(s - max)) ----
    float psum[4];
    for (int h = 0; h < H; ++h) psum[h] = 0.f;
    for (int i = t; i < dcount; i += bd) {
        int e = eidx[start + i];
        int src = (e < E) ? ei[e] : (e - E);
        for (int h = 0; h < H; ++h) {
            float s = asrc[(size_t)src * H + h] + ad_loc[h];
            s = (s > 0.f) ? s : NEG_SLOPE * s;
            psum[h] += __expf(s - shead[h]);
        }
    }
    for (int h = 0; h < H; ++h) {
        red[t] = psum[h];
        __syncthreads();
        for (int s2 = bd >> 1; s2 >= 1; s2 >>= 1) {
            if (t < s2) red[t] += red[t + s2];
            __syncthreads();
        }
        if (t == 0) shead[4 + h] = red[0];
        __syncthreads();
    }

    // ---- phase C: weighted accumulation of source features ----
    const int hd = t / C;
    const float mx = shead[hd];
    const float dn = shead[4 + hd] + 1e-16f;
    const float adh = ad_loc[hd];
    float acc = 0.f;
    for (int i = 0; i < dcount; ++i) {
        int e = eidx[start + i];
        int src = (e < E) ? ei[e] : (e - E);
        float s = asrc[(size_t)src * H + hd] + adh;
        s = (s > 0.f) ? s : NEG_SLOPE * s;
        float p = __expf(s - mx);
        acc += feat[(size_t)src * D + t] * p;
    }
    agg[(size_t)n * D + t] = acc / dn;
}

// ---------------- batchnorm stats (per-channel sum / sumsq) -----------------
__global__ void bn_stats_kernel(const float* __restrict__ agg, const float* __restrict__ bias,
                                float* __restrict__ gsum, float* __restrict__ gsumsq,
                                int Nn, int D, int rows_per_block) {
    const int t = threadIdx.x;
    int r0 = blockIdx.x * rows_per_block;
    int r1 = r0 + rows_per_block; if (r1 > Nn) r1 = Nn;
    float b = bias[t];
    float s = 0.f, s2 = 0.f;
    for (int r = r0; r < r1; ++r) {
        float v = agg[(size_t)r * D + t] + b;
        s += v; s2 += v * v;
    }
    atomicAdd(&gsum[t], s);
    atomicAdd(&gsumsq[t], s2);
}

// ---------------- batchnorm apply + ELU -> bf16 (layers 1,2) ----------------
__global__ void bn_apply_elu_bf16_kernel(const float* __restrict__ agg, const float* __restrict__ bias,
                                         const float* __restrict__ gsum, const float* __restrict__ gsumsq,
                                         const float* __restrict__ gamma, const float* __restrict__ beta,
                                         __bf16* __restrict__ outb, int Nn, int D) {
    size_t t = (size_t)blockIdx.x * blockDim.x + threadIdx.x;
    size_t n = (size_t)Nn * D;
    if (t >= n) return;
    int c = (int)(t % D);
    float invN = 1.0f / (float)Nn;
    float mu  = gsum[c] * invN;
    float var = gsumsq[c] * invN - mu * mu;
    float v = agg[t] + bias[c];
    float y = (v - mu) * rsqrtf(var + BN_EPS) * gamma[c] + beta[c];
    y = (y > 0.f) ? y : expm1f(y);
    outb[t] = (__bf16)y;
}

// ---------------- batchnorm apply (final layer, fp32, no ELU) ---------------
__global__ void bn_apply_final_kernel(const float* __restrict__ agg, const float* __restrict__ bias,
                                      const float* __restrict__ gsum, const float* __restrict__ gsumsq,
                                      const float* __restrict__ gamma, const float* __restrict__ beta,
                                      float* __restrict__ out, int Nn, int D) {
    size_t t = (size_t)blockIdx.x * blockDim.x + threadIdx.x;
    size_t n = (size_t)Nn * D;
    if (t >= n) return;
    int c = (int)(t % D);
    float invN = 1.0f / (float)Nn;
    float mu  = gsum[c] * invN;
    float var = gsumsq[c] * invN - mu * mu;
    float v = agg[t] + bias[c];
    out[t] = (v - mu) * rsqrtf(var + BN_EPS) * gamma[c] + beta[c];
}

// ===========================================================================
extern "C" void kernel_launch(void* const* d_in, const int* in_sizes, int n_in,
                              void* d_out, int out_size, void* d_ws, size_t ws_size,
                              hipStream_t stream) {
    (void)n_in; (void)out_size; (void)ws_size;

    constexpr int FIN = 128, HEADS = 4, OUTD = 64, D1 = 256, C = 64;
    const int N = in_sizes[0] / FIN;       // 50000
    const int E = in_sizes[1] / 2;         // 800000
    const int Etot = E + N;                // edges + self loops

    const float* x  = (const float*)d_in[0];
    const int*   ei = (const int*)d_in[1];
    const float* W[3]  = {(const float*)d_in[2],  (const float*)d_in[8],  (const float*)d_in[14]};
    const float* AS[3] = {(const float*)d_in[3],  (const float*)d_in[9],  (const float*)d_in[15]};
    const float* AD[3] = {(const float*)d_in[4],  (const float*)d_in[10], (const float*)d_in[16]};
    const float* B [3] = {(const float*)d_in[5],  (const float*)d_in[11], (const float*)d_in[17]};
    const float* G [3] = {(const float*)d_in[6],  (const float*)d_in[12], (const float*)d_in[18]};
    const float* BE[3] = {(const float*)d_in[7],  (const float*)d_in[13], (const float*)d_in[19]};

    // ---------------- workspace carve-up ----------------
    char* w = (char*)d_ws;
    size_t off = 0;
    auto carve = [&](size_t bytes) -> void* {
        void* p = w + off;
        off = (off + bytes + 255) & ~(size_t)255;
        return p;
    };
    float*  feat   = (float*) carve((size_t)N * D1 * 4);     // GEMM output (pre-aggregation h)
    float*  agg    = (float*) carve((size_t)N * D1 * 4);     // aggregated messages
    __bf16* hb     = (__bf16*)carve((size_t)N * D1 * 2);     // bf16 activations (GEMM input)
    __bf16* Wt     = (__bf16*)carve((size_t)D1 * D1 * 2);    // bf16 transposed weights
    float*  asrc   = (float*) carve((size_t)N * HEADS * 4);
    float*  adst   = (float*) carve((size_t)N * HEADS * 4);
    int*    deg    = (int*)   carve((size_t)N * 4);
    int*    fillp  = (int*)   carve((size_t)N * 4);
    int*    rowptr = (int*)   carve((size_t)N * 4);
    int*    eidx   = (int*)   carve((size_t)Etot * 4);
    int*    bsum   = (int*)   carve(SCAN_B * 4);
    float*  gsum   = (float*) carve(D1 * 4);
    float*  gsumsq = (float*) carve(D1 * 4);

    auto cdiv = [](size_t a, size_t b) { return (int)((a + b - 1) / b); };

    // ---------------- build CSR over destination nodes (once per launch) ----
    {
        fill_i32_kernel<<<cdiv((size_t)N, 256), 256, 0, stream>>>(deg, 0, (size_t)N);
        fill_i32_kernel<<<cdiv((size_t)N, 256), 256, 0, stream>>>(fillp, 0, (size_t)N);
        deg_count_kernel<<<cdiv((size_t)Etot, 256), 256, 0, stream>>>(ei, E, N, deg);
        int nb = cdiv((size_t)N, SCAN_B);
        scan_block_kernel<<<nb, SCAN_B, 0, stream>>>(deg, rowptr, bsum, N);
        scan_sums_kernel<<<1, SCAN_B, 0, stream>>>(bsum, nb);
        scan_add_kernel<<<cdiv((size_t)N, 256), 256, 0, stream>>>(rowptr, bsum, N);
        csr_scatter_kernel<<<cdiv((size_t)Etot, 256), 256, 0, stream>>>(ei, E, N, rowptr, fillp, eidx);
    }

    // layer 1 input: x -> bf16
    {
        size_t n = (size_t)N * FIN;
        f32_to_bf16_kernel<<<cdiv(n, 256), 256, 0, stream>>>(x, hb, n);
    }

    const int Din_[3]  = {FIN, D1, D1};
    const int Dout_[3] = {D1,  D1, OUTD};
    const int H_[3]    = {HEADS, HEADS, 1};

    for (int L = 0; L < 3; ++L) {
        const int Din = Din_[L], Dout = Dout_[L], H = H_[L];

        // weights: [Din, Dout] -> bf16 [Dout, Din]
        {
            size_t n = (size_t)Din * Dout;
            transpose_to_bf16_kernel<<<cdiv(n, 256), 256, 0, stream>>>(W[L], Wt, Din, Dout);
        }
        // GEMM: feat = hb @ W      (v_wmma_f32_16x16x32_bf16, B strip in LDS)
        {
            dim3 grid(cdiv((size_t)(N + 15) / 16, GEMM_WAVES), Dout / 64);
            wmma_gemm_kernel<<<grid, GEMM_WAVES * 32, 0, stream>>>(hb, Wt, feat, N, Din, Dout);
        }
        // attention logits per node/head
        alpha_kernel<<<N, Dout, 0, stream>>>(feat, AS[L], AD[L], asrc, adst, Dout, C);

        // fused segment-softmax + gather aggregation (atomic-free)
        gat_aggregate_csr_kernel<<<N, Dout, 0, stream>>>(ei, E, rowptr, deg, eidx,
                                                         feat, asrc, adst, agg, Dout, C, H);

        // batchnorm
        fill_f32_kernel<<<1, 256, 0, stream>>>(gsum, 0.f, (size_t)Dout);
        fill_f32_kernel<<<1, 256, 0, stream>>>(gsumsq, 0.f, (size_t)Dout);
        bn_stats_kernel<<<cdiv((size_t)N, 128), Dout, 0, stream>>>(agg, B[L], gsum, gsumsq, N, Dout, 128);
        size_t nel = (size_t)N * Dout;
        if (L < 2) {
            bn_apply_elu_bf16_kernel<<<cdiv(nel, 256), 256, 0, stream>>>(
                agg, B[L], gsum, gsumsq, G[L], BE[L], hb, N, Dout);
        } else {
            bn_apply_final_kernel<<<cdiv(nel, 256), 256, 0, stream>>>(
                agg, B[L], gsum, gsumsq, G[L], BE[L], (float*)d_out, N, Dout);
        }
    }
}
